// MultiChannelGNNEncoder_30812095382123
// MI455X (gfx1250) — matrix operand order repaired and checked by reading, not compiled, not run on hardware
//
#include <hip/hip_runtime.h>
#include <math.h>

// Problem constants (match reference setup_inputs)
#define NN   8192
#define DINc 1024
#define DD   512
#define EE   131072
#define KNNK 8
#define LAYERS 2

typedef __bf16 bf16;
typedef __attribute__((ext_vector_type(16))) __bf16 bf16x16;
typedef __attribute__((ext_vector_type(8)))  float  f32x8;
typedef __attribute__((ext_vector_type(4)))  float  f32x4;
typedef __attribute__((ext_vector_type(4)))  unsigned int u32x4;

union Frag { bf16x16 v; u32x4 q[2]; };

__device__ __forceinline__ float relu6f(float v){ return fminf(fmaxf(v, 0.f), 6.f); }
__device__ __forceinline__ float lrelu(float v){ return v > 0.f ? v : 0.2f * v; }
// monotone float<->uint encoding for atomicMax-based segment max
__device__ __forceinline__ unsigned encf(float x){
  unsigned u = __float_as_uint(x);
  return (u & 0x80000000u) ? ~u : (u | 0x80000000u);
}
__device__ __forceinline__ float decf(unsigned u){
  return (u & 0x80000000u) ? __uint_as_float(u ^ 0x80000000u) : __uint_as_float(~u);
}

// ---------------- conversion kernels ----------------
__global__ __launch_bounds__(256) void cvt_f32_bf16(const float* __restrict__ in,
                                                    bf16* __restrict__ out, int n) {
  int i = blockIdx.x * 256 + threadIdx.x;
  if (i < n) out[i] = (bf16)in[i];
}

// W[k][n] f32 -> Wt[n][k] bf16
__global__ __launch_bounds__(256) void cvtT_f32_bf16(const float* __restrict__ W,
                                                     bf16* __restrict__ Wt, int K, int N) {
  int i = blockIdx.x * 256 + threadIdx.x;
  if (i < K * N) {
    int k = i / N, n = i % N;
    Wt[(size_t)n * K + k] = (bf16)W[i];
  }
}

// ---------------- WMMA GEMM: C = act(A[M,K] @ Bt[N,K]^T + bias) ----------------
// block 256 threads = 8 waves, 128x128 tile, BK=32, per-wave 64x32 (4x2 WMMA tiles)
// Software-pipelined: global loads for tile k+1 in flight during WMMA on tile k;
// double-buffered LDS, one barrier per K-step.
template<int ACT>
__global__ __launch_bounds__(256) void gemm_bf16(const bf16* __restrict__ A,
                                                 const bf16* __restrict__ Bt,
                                                 const float* __restrict__ bias,
                                                 float* __restrict__ Cf,
                                                 bf16* __restrict__ Cb,
                                                 int M, int Nc, int K) {
  __shared__ bf16 As[2][128 * 40];
  __shared__ bf16 Bs[2][128 * 40];
  const int tid = threadIdx.x;
  const int lane = tid & 31, wid = tid >> 5;
  const int bm = blockIdx.x * 128, bn = blockIdx.y * 128;
  const int waveM = wid >> 2, waveN = wid & 3;
  const int r = tid >> 1, half = tid & 1;
  const int kbA = (lane >> 4) * 8;    // A fragment K base (split at 16)
  const int kbB = (lane >> 4) * 8;    // B fragment K base (split at 16)
  f32x8 c[4][2] = {};
  (void)M;

  const bf16* gA = A + (size_t)(bm + r) * K + half * 16;
  const bf16* gB = Bt + (size_t)(bn + r) * K + half * 16;
  const int lofs = r * 40 + half * 16;

  // preload tile 0 into LDS buffer 0
  u32x4 ra0 = ((const u32x4*)gA)[0], ra1 = ((const u32x4*)gA)[1];
  u32x4 rb0 = ((const u32x4*)gB)[0], rb1 = ((const u32x4*)gB)[1];
  {
    u32x4* la = (u32x4*)(As[0] + lofs); la[0] = ra0; la[1] = ra1;
    u32x4* lb = (u32x4*)(Bs[0] + lofs); lb[0] = rb0; lb[1] = rb1;
  }
  __syncthreads();

  int cur = 0;
  for (int k0 = 0; k0 < K; k0 += 32) {
    const bool more = (k0 + 32 < K);
    if (more) {  // issue next tile's global loads before the math
      const u32x4* ga = (const u32x4*)(gA + k0 + 32);
      ra0 = ga[0]; ra1 = ga[1];
      const u32x4* gb = (const u32x4*)(gB + k0 + 32);
      rb0 = gb[0]; rb1 = gb[1];
    }
    Frag a[4], b[2];
#pragma unroll
    for (int mi = 0; mi < 4; ++mi) {
      const bf16* p = As[cur] + (waveM * 64 + mi * 16 + (lane & 15)) * 40 + kbA;
      a[mi].q[0] = *(const u32x4*)p;
      a[mi].q[1] = *(const u32x4*)(p + 16);
    }
#pragma unroll
    for (int ni = 0; ni < 2; ++ni) {
      const bf16* p = Bs[cur] + (waveN * 32 + ni * 16 + (lane & 15)) * 40 + kbB;
      b[ni].q[0] = *(const u32x4*)p;
      b[ni].q[1] = *(const u32x4*)(p + 16);
    }
#pragma unroll
    for (int mi = 0; mi < 4; ++mi)
#pragma unroll
      for (int ni = 0; ni < 2; ++ni)
        c[mi][ni] = __builtin_amdgcn_wmma_f32_16x16x32_bf16(
            false, a[mi].v, false, b[ni].v, (short)0, c[mi][ni], false, false);
    if (more) {  // stash prefetched tile into the other buffer
      u32x4* la = (u32x4*)(As[cur ^ 1] + lofs); la[0] = ra0; la[1] = ra1;
      u32x4* lb = (u32x4*)(Bs[cur ^ 1] + lofs); lb[0] = rb0; lb[1] = rb1;
      __syncthreads();
      cur ^= 1;
    }
  }

  const int mloc = (lane >> 4) << 3;
#pragma unroll
  for (int mi = 0; mi < 4; ++mi) {
#pragma unroll
    for (int ni = 0; ni < 2; ++ni) {
      const int col = bn + waveN * 32 + ni * 16 + (lane & 15);
      const float bv = bias ? bias[col] : 0.f;
#pragma unroll
      for (int v = 0; v < 8; ++v) {
        const int row = bm + waveM * 64 + mi * 16 + mloc + v;
        float val = c[mi][ni][v] + bv;
        if (ACT == 1) val = relu6f(val);
        const size_t idx = (size_t)row * Nc + col;
        if (Cf) Cf[idx] = val;
        if (Cb) Cb[idx] = (bf16)val;
      }
    }
  }
}

// ---------------- row sum-of-squares (for KNN distances) ----------------
__global__ __launch_bounds__(256) void row_sumsq(const float* __restrict__ h,
                                                 float* __restrict__ sq) {
  const int row = blockIdx.x * 8 + (threadIdx.x >> 5);
  const int lane = threadIdx.x & 31;
  const f32x4* h4 = (const f32x4*)(h + (size_t)row * DD);
  float p = 0.f;
#pragma unroll
  for (int j = 0; j < 4; ++j) {
    f32x4 v = h4[lane + j * 32];
    p += v.x * v.x + v.y * v.y + v.z * v.z + v.w * v.w;
  }
#pragma unroll
  for (int off = 16; off; off >>= 1) p += __shfl_down(p, off);
  if (lane == 0) sq[row] = p;
}

// ---------------- fused KNN top-8: WMMA Gram tiles + register top-k ----------------
// block = 64 threads (2 waves), 32 rows per block; streams 16-col tiles over all cols.
// A strip (16x512) register-resident; B fragments prefetched one K-step ahead
// (and across the top-k scan at tile boundaries) so loads overlap WMMA.
__global__ __launch_bounds__(64) void knn_topk(const bf16* __restrict__ hb,
                                               const float* __restrict__ sq,
                                               int* __restrict__ knn) {
  __shared__ float dist[32][17];
  const int tid = threadIdx.x, lane = tid & 31, wid = tid >> 5;
  const int bm = blockIdx.x * 32;
  const int m0 = bm + wid * 16;
  const int kbA = (lane >> 4) * 8;
  const int kbB = (lane >> 4) * 8;
  const int nlane = lane & 15;
  const bf16* arow = hb + (size_t)(m0 + nlane) * DD;

  // register-resident A fragments for this wave's 16-row strip (K = 512)
  Frag afr[16];
#pragma unroll
  for (int ks = 0; ks < 16; ++ks) {
    const bf16* pa = arow + ks * 32;
    afr[ks].q[0] = *(const u32x4*)(pa + kbA);
    afr[ks].q[1] = *(const u32x4*)(pa + kbA + 16);
  }

  float bd[8]; int bi[8];
#pragma unroll
  for (int j = 0; j < 8; ++j) { bd[j] = 3.0e38f; bi[j] = 0; }
  float worst = 3.0e38f; int wslot = 0;
  const int myrow = bm + tid;

  // B fragment prefetch pipeline
  Frag bn;
  {
    const bf16* p = hb + (size_t)nlane * DD + kbB;
    bn.q[0] = *(const u32x4*)p;
    bn.q[1] = *(const u32x4*)(p + 16);
  }
  for (int ct = 0; ct < NN / 16; ++ct) {
    f32x8 c0 = {}, c1 = {};
#pragma unroll
    for (int ks = 0; ks < 16; ++ks) {
      Frag bc = bn;
      // prefetch next fragment (next K-step, or K-step 0 of the next column tile)
      const int nct = (ks == 15) ? ((ct + 1) & (NN / 16 - 1)) : ct;
      const int nks = (ks == 15) ? 0 : ks + 1;
      const bf16* p = hb + (size_t)(nct * 16 + nlane) * DD + nks * 32 + kbB;
      bn.q[0] = *(const u32x4*)p;
      bn.q[1] = *(const u32x4*)(p + 16);
      if (ks & 1)
        c1 = __builtin_amdgcn_wmma_f32_16x16x32_bf16(false, afr[ks].v, false, bc.v,
                                                     (short)0, c1, false, false);
      else
        c0 = __builtin_amdgcn_wmma_f32_16x16x32_bf16(false, afr[ks].v, false, bc.v,
                                                     (short)0, c0, false, false);
    }
#pragma unroll
    for (int v = 0; v < 8; ++v)
      dist[wid * 16 + ((lane >> 4) << 3) + v][nlane] = c0[v] + c1[v];
    __syncthreads();
    if (tid < 32) {
#pragma unroll
      for (int n = 0; n < 16; ++n) {
        const int col = ct * 16 + n;
        const float d = sq[col] - 2.f * dist[tid][n];
        if (col != myrow && d < worst) {
          bd[wslot] = d; bi[wslot] = col;
          worst = bd[0]; wslot = 0;
#pragma unroll
          for (int j = 1; j < 8; ++j)
            if (bd[j] > worst) { worst = bd[j]; wslot = j; }
        }
      }
    }
    __syncthreads();
  }
  if (tid < 32) {
#pragma unroll
    for (int j = 0; j < 8; ++j) knn[(size_t)myrow * KNNK + j] = bi[j];
  }
}

// ---------------- spatial-channel GAT (given edge list) ----------------
// logits[e] = att . leaky_relu(xl[src] + xr[tgt] + ea[e]*We)   (wave per edge)
__global__ __launch_bounds__(256) void edge_logits_sp(const float* __restrict__ xl,
                                                      const float* __restrict__ xr,
                                                      const float* __restrict__ ea,
                                                      const float* __restrict__ We,
                                                      const float* __restrict__ att,
                                                      const int* __restrict__ src,
                                                      const int* __restrict__ tgt,
                                                      float* __restrict__ logits) {
  const int e = blockIdx.x * 8 + (threadIdx.x >> 5);
  const int lane = threadIdx.x & 31;
  const int s = src[e], t = tgt[e];
  const float av = ea[e];
  const f32x4* xls = (const f32x4*)(xl + (size_t)s * DD);
  const f32x4* xrt = (const f32x4*)(xr + (size_t)t * DD);
  const f32x4* We4 = (const f32x4*)We;
  const f32x4* at4 = (const f32x4*)att;
  float p = 0.f;
#pragma unroll
  for (int j = 0; j < 4; ++j) {
    const int idx = lane + j * 32;
    f32x4 v1 = xls[idx], v2 = xrt[idx], w = We4[idx], a = at4[idx];
    p += a.x * lrelu(v1.x + v2.x + av * w.x);
    p += a.y * lrelu(v1.y + v2.y + av * w.y);
    p += a.z * lrelu(v1.z + v2.z + av * w.z);
    p += a.w * lrelu(v1.w + v2.w + av * w.w);
  }
#pragma unroll
  for (int off = 16; off; off >>= 1) p += __shfl_down(p, off);
  if (lane == 0) logits[e] = p;
}

__global__ __launch_bounds__(256) void seg_max(const float* __restrict__ logits,
                                               const int* __restrict__ tgt,
                                               unsigned* __restrict__ mkey) {
  const int i = blockIdx.x * 256 + threadIdx.x;
  atomicMax(&mkey[tgt[i]], encf(logits[i]));
}

__global__ __launch_bounds__(256) void seg_expsum(const float* __restrict__ logits,
                                                  const int* __restrict__ tgt,
                                                  const unsigned* __restrict__ mkey,
                                                  float* __restrict__ expv,
                                                  float* __restrict__ ssum) {
  const int i = blockIdx.x * 256 + threadIdx.x;
  const float m = decf(mkey[tgt[i]]);
  const float v = expf(logits[i] - m);
  expv[i] = v;
  __hip_atomic_fetch_add(&ssum[tgt[i]], v, __ATOMIC_RELAXED, __HIP_MEMORY_SCOPE_AGENT);
}

// out[tgt] += alpha * xl[src]   (block of 128 per edge, 4 floats/thread)
__global__ __launch_bounds__(128) void scatter_sp(const float* __restrict__ expv,
                                                  const float* __restrict__ ssum,
                                                  const int* __restrict__ src,
                                                  const int* __restrict__ tgt,
                                                  const float* __restrict__ xl,
                                                  float* __restrict__ out) {
  const int e = blockIdx.x;
  const int t = tgt[e], s = src[e];
  const float alpha = expv[e] / ssum[t];
  const f32x4 v = ((const f32x4*)(xl + (size_t)s * DD))[threadIdx.x];
  float* ob = out + (size_t)t * DD + threadIdx.x * 4;
  __hip_atomic_fetch_add(ob + 0, alpha * v.x, __ATOMIC_RELAXED, __HIP_MEMORY_SCOPE_AGENT);
  __hip_atomic_fetch_add(ob + 1, alpha * v.y, __ATOMIC_RELAXED, __HIP_MEMORY_SCOPE_AGENT);
  __hip_atomic_fetch_add(ob + 2, alpha * v.z, __ATOMIC_RELAXED, __HIP_MEMORY_SCOPE_AGENT);
  __hip_atomic_fetch_add(ob + 3, alpha * v.w, __ATOMIC_RELAXED, __HIP_MEMORY_SCOPE_AGENT);
}

// ---------------- latent-channel GAT: 8 contiguous KNN edges per node, local softmax ----------------
__global__ __launch_bounds__(128) void latent_gat(const float* __restrict__ xl,
                                                  const float* __restrict__ xr,
                                                  const float* __restrict__ att,
                                                  const float* __restrict__ bias,
                                                  const int* __restrict__ knn,
                                                  float* __restrict__ out) {
  __shared__ float red[KNNK][4];
  __shared__ float lg[KNNK];
  const int i = blockIdx.x, tid = threadIdx.x;
  const int lane = tid & 31, w = tid >> 5;
  const f32x4 xrv = ((const f32x4*)(xr + (size_t)i * DD))[tid];
  const f32x4 attv = ((const f32x4*)att)[tid];
  f32x4 xlv[KNNK]; float part[KNNK];
#pragma unroll
  for (int k = 0; k < KNNK; ++k) {
    const int nb = knn[i * KNNK + k];
    xlv[k] = ((const f32x4*)(xl + (size_t)nb * DD))[tid];
    part[k] = attv.x * lrelu(xlv[k].x + xrv.x) + attv.y * lrelu(xlv[k].y + xrv.y) +
              attv.z * lrelu(xlv[k].z + xrv.z) + attv.w * lrelu(xlv[k].w + xrv.w);
  }
#pragma unroll
  for (int k = 0; k < KNNK; ++k) {
    float p = part[k];
#pragma unroll
    for (int off = 16; off; off >>= 1) p += __shfl_down(p, off);
    if (lane == 0) red[k][w] = p;
  }
  __syncthreads();
  if (tid < KNNK) lg[tid] = red[tid][0] + red[tid][1] + red[tid][2] + red[tid][3];
  __syncthreads();
  float m = -3.0e38f;
#pragma unroll
  for (int k = 0; k < KNNK; ++k) m = fmaxf(m, lg[k]);
  float a[KNNK]; float s = 0.f;
#pragma unroll
  for (int k = 0; k < KNNK; ++k) { a[k] = expf(lg[k] - m); s += a[k]; }
  const float inv = 1.f / s;
  f32x4 o = ((const f32x4*)bias)[tid];
#pragma unroll
  for (int k = 0; k < KNNK; ++k) {
    const float ak = a[k] * inv;
    o.x += ak * xlv[k].x; o.y += ak * xlv[k].y;
    o.z += ak * xlv[k].z; o.w += ak * xlv[k].w;
  }
  ((f32x4*)(out + (size_t)i * DD))[tid] = o;
}

// h = relu6(sp_accum + bias_sp + lat); also refresh bf16 copy of h
__global__ __launch_bounds__(256) void fuse_relu6(const float* __restrict__ sp,
                                                  const float* __restrict__ bias_sp,
                                                  const float* __restrict__ lat,
                                                  float* __restrict__ h,
                                                  bf16* __restrict__ hbb) {
  const int i = blockIdx.x * 256 + threadIdx.x;
  const float v = relu6f(sp[i] + bias_sp[i % DD] + lat[i]);
  h[i] = v;
  hbb[i] = (bf16)v;
}

// ---------------- host orchestration ----------------
extern "C" void kernel_launch(void* const* d_in, const int* in_sizes, int n_in,
                              void* d_out, int out_size, void* d_ws, size_t ws_size,
                              hipStream_t stream) {
  (void)in_sizes; (void)n_in; (void)out_size; (void)ws_size;
  const float* x       = (const float*)d_in[0];
  const float* ea      = (const float*)d_in[1];
  const float* W0      = (const float*)d_in[2];
  const float* b0      = (const float*)d_in[3];
  const float* W1      = (const float*)d_in[4];
  const float* b1      = (const float*)d_in[5];
  const float* W2      = (const float*)d_in[6];
  const float* b2      = (const float*)d_in[7];
  const float* Wl_sp   = (const float*)d_in[8];
  const float* bl_sp   = (const float*)d_in[9];
  const float* Wr_sp   = (const float*)d_in[10];
  const float* br_sp   = (const float*)d_in[11];
  const float* att_sp  = (const float*)d_in[12];
  const float* bias_sp = (const float*)d_in[13];
  const float* We_sp   = (const float*)d_in[14];
  const float* Wl_lat  = (const float*)d_in[15];
  const float* bl_lat  = (const float*)d_in[16];
  const float* Wr_lat  = (const float*)d_in[17];
  const float* br_lat  = (const float*)d_in[18];
  const float* att_lat = (const float*)d_in[19];
  const float* bias_lat= (const float*)d_in[20];
  const int*   eidx    = (const int*)d_in[21];
  const int* srcp = eidx;
  const int* tgtp = eidx + EE;
  float* h = (float*)d_out;  // persistent node features live in d_out

  // bump allocator on d_ws
  size_t cur = 0;
  char* wsb = (char*)d_ws;
  auto alloc = [&](size_t bytes) -> void* {
    cur = (cur + 255) & ~(size_t)255;
    void* p = (void*)(wsb + cur);
    cur += bytes;
    return p;
  };
  // persistent region
  bf16* W0t = (bf16*)alloc((size_t)DINc * 512 * 2);   // [512][1024]
  bf16* W1t = (bf16*)alloc((size_t)512 * 256 * 2);    // [256][512]
  bf16* W2t = (bf16*)alloc((size_t)256 * DD * 2);     // [512][256]
  bf16* Wt[LAYERS][4];
  for (int l = 0; l < LAYERS; ++l)
    for (int j = 0; j < 4; ++j) Wt[l][j] = (bf16*)alloc((size_t)DD * DD * 2);
  bf16* hb = (bf16*)alloc((size_t)NN * DD * 2);
  const size_t scratch0 = cur;
  // encoder scratch
  bf16* xb  = (bf16*)alloc((size_t)NN * DINc * 2);
  bf16* h1b = (bf16*)alloc((size_t)NN * 512 * 2);
  bf16* h2b = (bf16*)alloc((size_t)NN * 256 * 2);
  // layer scratch (aliases encoder scratch; temporally disjoint)
  cur = scratch0;
  float* xl_sp  = (float*)alloc((size_t)NN * DD * 4);
  float* xr_sp  = (float*)alloc((size_t)NN * DD * 4);
  float* xl_lat = (float*)alloc((size_t)NN * DD * 4);
  float* xr_lat = (float*)alloc((size_t)NN * DD * 4);
  float* out_sp = (float*)alloc((size_t)NN * DD * 4);
  float* out_lat= (float*)alloc((size_t)NN * DD * 4);
  float* sq     = (float*)alloc((size_t)NN * 4);
  int*   knn    = (int*)alloc((size_t)NN * KNNK * 4);
  float* logits = (float*)alloc((size_t)EE * 4);
  float* expv   = (float*)alloc((size_t)EE * 4);
  unsigned* mkey= (unsigned*)alloc((size_t)NN * 4);
  float* ssum   = (float*)alloc((size_t)NN * 4);

  // ---- weight conversion (bf16, transposed to [N][K]) ----
  cvtT_f32_bf16<<<(DINc * 512 + 255) / 256, 256, 0, stream>>>(W0, W0t, DINc, 512);
  cvtT_f32_bf16<<<(512 * 256 + 255) / 256, 256, 0, stream>>>(W1, W1t, 512, 256);
  cvtT_f32_bf16<<<(256 * DD + 255) / 256, 256, 0, stream>>>(W2, W2t, 256, DD);
  for (int l = 0; l < LAYERS; ++l) {
    const size_t o = (size_t)l * DD * DD;
    cvtT_f32_bf16<<<(DD * DD + 255) / 256, 256, 0, stream>>>(Wl_sp + o,  Wt[l][0], DD, DD);
    cvtT_f32_bf16<<<(DD * DD + 255) / 256, 256, 0, stream>>>(Wr_sp + o,  Wt[l][1], DD, DD);
    cvtT_f32_bf16<<<(DD * DD + 255) / 256, 256, 0, stream>>>(Wl_lat + o, Wt[l][2], DD, DD);
    cvtT_f32_bf16<<<(DD * DD + 255) / 256, 256, 0, stream>>>(Wr_lat + o, Wt[l][3], DD, DD);
  }

  // ---- encoder MLP: relu6(xW0+b0) -> relu6(·W1+b1) -> ·W2+b2 ----
  cvt_f32_bf16<<<(NN * DINc + 255) / 256, 256, 0, stream>>>(x, xb, NN * DINc);
  gemm_bf16<1><<<dim3(NN / 128, 512 / 128), 256, 0, stream>>>(
      xb, W0t, b0, nullptr, h1b, NN, 512, DINc);
  gemm_bf16<1><<<dim3(NN / 128, 256 / 128), 256, 0, stream>>>(
      h1b, W1t, b1, nullptr, h2b, NN, 256, 512);
  gemm_bf16<0><<<dim3(NN / 128, DD / 128), 256, 0, stream>>>(
      h2b, W2t, b2, h, hb, NN, DD, 256);

  // ---- GNN layers ----
  for (int l = 0; l < LAYERS; ++l) {
    const size_t ov = (size_t)l * DD;
    // 4 dense transforms of h
    gemm_bf16<0><<<dim3(NN / 128, DD / 128), 256, 0, stream>>>(
        hb, Wt[l][0], bl_sp + ov, xl_sp, nullptr, NN, DD, DD);
    gemm_bf16<0><<<dim3(NN / 128, DD / 128), 256, 0, stream>>>(
        hb, Wt[l][1], br_sp + ov, xr_sp, nullptr, NN, DD, DD);
    gemm_bf16<0><<<dim3(NN / 128, DD / 128), 256, 0, stream>>>(
        hb, Wt[l][2], bl_lat + ov, xl_lat, nullptr, NN, DD, DD);
    gemm_bf16<0><<<dim3(NN / 128, DD / 128), 256, 0, stream>>>(
        hb, Wt[l][3], br_lat + ov, xr_lat, nullptr, NN, DD, DD);
    // fresh KNN graph on h
    row_sumsq<<<NN / 8, 256, 0, stream>>>(h, sq);
    knn_topk<<<NN / 32, 64, 0, stream>>>(hb, sq, knn);
    // spatial channel: logits -> segment softmax -> scatter
    edge_logits_sp<<<EE / 8, 256, 0, stream>>>(xl_sp, xr_sp, ea, We_sp + ov,
                                               att_sp + ov, srcp, tgtp, logits);
    hipMemsetAsync(mkey, 0, (size_t)NN * 4, stream);
    hipMemsetAsync(ssum, 0, (size_t)NN * 4, stream);
    hipMemsetAsync(out_sp, 0, (size_t)NN * DD * 4, stream);
    seg_max<<<EE / 256, 256, 0, stream>>>(logits, tgtp, mkey);
    seg_expsum<<<EE / 256, 256, 0, stream>>>(logits, tgtp, mkey, expv, ssum);
    scatter_sp<<<EE, 128, 0, stream>>>(expv, ssum, srcp, tgtp, xl_sp, out_sp);
    // latent channel: per-node local softmax over its 8 KNN edges
    latent_gat<<<NN, 128, 0, stream>>>(xl_lat, xr_lat, att_lat + ov, bias_lat + ov,
                                       knn, out_lat);
    // fuse channels
    fuse_relu6<<<(NN * DD) / 256, 256, 0, stream>>>(out_sp, bias_sp + ov, out_lat, h, hb);
  }
}
